// PairwiseAttentionStructModule_53137335386463
// MI455X (gfx1250) — matrix operand order repaired and checked by reading, not compiled, not run on hardware
//
#include <hip/hip_runtime.h>

// Dims (fixed by the reference)
#define BB   8
#define NNV  1024
#define MML  64
#define HH   128
#define HHF  64      // H/2

typedef __attribute__((ext_vector_type(16))) _Float16 v16h;
typedef __attribute__((ext_vector_type(8)))  _Float16 v8h;
typedef __attribute__((ext_vector_type(8)))  float    v8f;
typedef __attribute__((ext_vector_type(4)))  float    vf4;
typedef __attribute__((ext_vector_type(4)))  int      v4i;

typedef __attribute__((address_space(1))) v4i gv4i;   // global int4
typedef __attribute__((address_space(3))) v4i lv4i;   // LDS int4

union AFrag { v16h v; _Float16 h[16]; };

#if __has_builtin(__builtin_amdgcn_global_load_async_to_lds_b128) && \
    __has_builtin(__builtin_amdgcn_s_wait_asynccnt)
#define K2_ASYNC 1
#else
#define K2_ASYNC 0
#endif

#if K2_ASYNC
#define ASYNC_WAIT(n_) do { __builtin_amdgcn_s_wait_asynccnt(n_); asm volatile("" ::: "memory"); } while (0)
#else
#define ASYNC_WAIT(n_) do { } while (0)
#endif

// ---------------------------------------------------------------------------
// Kernel 1: v_part = (Xv@Wv+bv)@W1[H:], l_part = (Xl@Wl+bl)@W1[:H]
// One wave per 16-row band. Bands 0..511 = virtual, 512..543 = ligand.
// ---------------------------------------------------------------------------
__global__ __launch_bounds__(256) void k1_project(
    const float* __restrict__ Xv, const float* __restrict__ Xl,
    const float* __restrict__ Wv, const float* __restrict__ bv,
    const float* __restrict__ Wl, const float* __restrict__ bl,
    const float* __restrict__ W1,
    float* __restrict__ ws_v, float* __restrict__ ws_l)
{
    __shared__ __attribute__((aligned(32))) _Float16 WS[4 * 8 * 32 * 16]; // 32 KB (reused Wa->Wb)
    __shared__ __attribute__((aligned(32))) _Float16 Ys[8][16 * 128];     // 32 KB (per-wave Y)

    const int lane = threadIdx.x & 31;
    const int wave = threadIdx.x >> 5;
    const int band = blockIdx.x * 8 + wave;
    const bool lig = band >= 512;

    const float* X  = lig ? Xl : Xv;
    const float* Wa = lig ? Wl : Wv;
    const float* ba = lig ? bl : bv;
    const float* Wb = lig ? W1 : (W1 + HH * HH);   // W1[:H] vs W1[H:]
    float* outp     = lig ? ws_l : ws_v;
    const int rowbase = (lig ? (band - 512) : band) * 16;

    // ---- stage Wa (B-swizzled f16) ----
    for (int i = threadIdx.x; i < 4 * 8 * 32 * 16; i += 256) {
        const int hh = i & 15, ln = (i >> 4) & 31, st = i >> 9;
        const int s = st >> 3, t = st & 7;
        const int k = s * 32 + (hh & 8) * 2 + ((ln >= 16) ? 8 : 0) + (hh & 7);
        const int n = t * 16 + (ln & 15);
        WS[i] = (_Float16)Wa[k * HH + n];
    }
    __syncthreads();

    const int m  = lane & 15;
    const int hi = lane >> 4;

    // ---- A fragments from X (f32 global -> f16), doc 16-bit A layout ----
    AFrag a[4];
#pragma unroll
    for (int s = 0; s < 4; ++s) {
        const int c = s * 32 + hi * 8;
        const float* xr = X + (size_t)(rowbase + m) * HH;
        vf4 g0 = *(const vf4*)(xr + c);
        vf4 g1 = *(const vf4*)(xr + c + 4);
        vf4 g2 = *(const vf4*)(xr + c + 16);
        vf4 g3 = *(const vf4*)(xr + c + 20);
#pragma unroll
        for (int j = 0; j < 4; ++j) {
            a[s].h[j]      = (_Float16)g0[j];
            a[s].h[4 + j]  = (_Float16)g1[j];
            a[s].h[8 + j]  = (_Float16)g2[j];
            a[s].h[12 + j] = (_Float16)g3[j];
        }
    }

    // ---- step A: Y = X16 @ Wa + ba  -> LDS row-major f16 ----
#pragma unroll
    for (int t = 0; t < 8; ++t) {
        v8f acc = {};
#pragma unroll
        for (int s = 0; s < 4; ++s) {
            const v16h bfr = *(const v16h*)&WS[((s * 8 + t) * 32 + lane) * 16];
            acc = __builtin_amdgcn_wmma_f32_16x16x32_f16(
                false, a[s].v, false, bfr, (short)0, acc, false, false);
        }
        const float bias = ba[t * 16 + m];
#pragma unroll
        for (int r = 0; r < 8; ++r)
            Ys[wave][(r + hi * 8) * 128 + t * 16 + m] = (_Float16)(acc[r] + bias);
    }
    __syncthreads();

    // ---- restage Wb into the same LDS buffer ----
    for (int i = threadIdx.x; i < 4 * 8 * 32 * 16; i += 256) {
        const int hh = i & 15, ln = (i >> 4) & 31, st = i >> 9;
        const int s = st >> 3, t = st & 7;
        const int k = s * 32 + (hh & 8) * 2 + ((ln >= 16) ? 8 : 0) + (hh & 7);
        const int n = t * 16 + (ln & 15);
        WS[i] = (_Float16)Wb[k * HH + n];
    }
    __syncthreads();

    // ---- step B: out = Y @ Wb ----
    AFrag a2[4];
#pragma unroll
    for (int s = 0; s < 4; ++s) {
        const int c = s * 32 + hi * 8;
        v8h g0 = *(const v8h*)&Ys[wave][m * 128 + c];
        v8h g1 = *(const v8h*)&Ys[wave][m * 128 + c + 16];
#pragma unroll
        for (int j = 0; j < 8; ++j) { a2[s].h[j] = g0[j]; a2[s].h[8 + j] = g1[j]; }
    }
#pragma unroll
    for (int t = 0; t < 8; ++t) {
        v8f acc = {};
#pragma unroll
        for (int s = 0; s < 4; ++s) {
            const v16h bfr = *(const v16h*)&WS[((s * 8 + t) * 32 + lane) * 16];
            acc = __builtin_amdgcn_wmma_f32_16x16x32_f16(
                false, a2[s].v, false, bfr, (short)0, acc, false, false);
        }
#pragma unroll
        for (int r = 0; r < 8; ++r)
            outp[(size_t)(rowbase + r + hi * 8) * HH + t * 16 + m] = acc[r];
    }
}

// ---------------------------------------------------------------------------
// Kernel 2: z[b,m,n] = relu( relu(l_part[m]+v_part[n]+b1) @ W2 + b2 ) @ W3 + b3
// Block = (b, 16-row m tile); 8 waves, each covering 128 n.
// W2 WMMA-B fragments live in registers (loop-invariant); v_part tiles are
// double-buffered in LDS via GLOBAL_LOAD_ASYNC_TO_LDS_B128 (ASYNCcnt-gated).
// ---------------------------------------------------------------------------
__global__ __launch_bounds__(256) void k2_pair_z(
    const float* __restrict__ ws_v, const float* __restrict__ ws_l,
    const float* __restrict__ b1,   const float* __restrict__ W2,
    const float* __restrict__ b2,   const float* __restrict__ W3,
    const float* __restrict__ b3,   float* __restrict__ ws_z)
{
    __shared__ __attribute__((aligned(32))) float    lpb[16 * 128];          //   8 KB: l_part + b1
    __shared__ __attribute__((aligned(32))) _Float16 W2S[4 * 4 * 32 * 16];   //  16 KB: swizzled W2
    __shared__ __attribute__((aligned(16))) float    VS[8][2][16 * 128];     // 128 KB: per-wave dbl buf

    const int lane = threadIdx.x & 31;
    const int wave = threadIdx.x >> 5;
    const int b    = blockIdx.x >> 2;
    const int m0   = (blockIdx.x & 3) * 16;

    // stage l_part tile with b1 folded in
    for (int i = threadIdx.x; i < 16 * 128; i += 256) {
        const int mm = i >> 7, k = i & 127;
        lpb[i] = ws_l[(size_t)(b * MML + m0 + mm) * HH + k] + b1[k];
    }
    // stage W2 (B-swizzled f16) -- read once into registers after the barrier
    for (int i = threadIdx.x; i < 4 * 4 * 32 * 16; i += 256) {
        const int hh = i & 15, ln = (i >> 4) & 31, st = i >> 9;
        const int s = st >> 2, t = st & 3;
        const int k = s * 32 + (hh & 8) * 2 + ((ln >= 16) ? 8 : 0) + (hh & 7);
        const int n = t * 16 + (ln & 15);
        W2S[i] = (_Float16)W2[k * HHF + n];
    }
    __syncthreads();

    const int m  = lane & 15;
    const int hi = lane >> 4;

    // hoist all 16 W2 B-fragments into registers (invariant over the n loop)
    v16h w2f[16];
#pragma unroll
    for (int f = 0; f < 16; ++f)
        w2f[f] = *(const v16h*)&W2S[(f * 32 + lane) * 16];

    float b2r[4], w3r[4];
#pragma unroll
    for (int t = 0; t < 4; ++t) { b2r[t] = b2[t * 16 + m]; w3r[t] = W3[t * 16 + m]; }
    const float b3v = b3[0];

    // async-stage one 16-row v_part tile (8 KB) into VS[wave][buf]
    auto issue_tile = [&](int nt, int buf) {
        const size_t base = (size_t)(b * NNV + wave * 128 + nt * 16) * HH;
#if K2_ASYNC
#pragma unroll
        for (int i = 0; i < 16; ++i) {
            const int e = (i * 32 + lane) * 4;   // float offset, 16B per lane per op
            __builtin_amdgcn_global_load_async_to_lds_b128(
                (gv4i*)(ws_v + base + e),
                (lv4i*)&VS[wave][buf][e],
                0, 0);
        }
#else
        for (int i = lane; i < 16 * 128; i += 32)
            VS[wave][buf][i] = ws_v[base + i];
#endif
    };

    issue_tile(0, 0);
    for (int nt = 0; nt < 8; ++nt) {
        const int buf = nt & 1;
        if (nt < 7) { issue_tile(nt + 1, buf ^ 1); ASYNC_WAIT(16); }
        else        { ASYNC_WAIT(0); }

        const int n0 = wave * 128 + nt * 16;
        for (int nn = 0; nn < 16; ++nn) {
            // A fragments: h[m,k] = relu(lpb[m,k] + v[n,k])
            AFrag a[4];
            const float* lr = &lpb[m * 128];
            const float* vr = &VS[wave][buf][nn * 128];
#pragma unroll
            for (int s = 0; s < 4; ++s) {
                const int c = s * 32 + hi * 8;
                vf4 l0 = *(const vf4*)(lr + c);
                vf4 l1 = *(const vf4*)(lr + c + 4);
                vf4 l2 = *(const vf4*)(lr + c + 16);
                vf4 l3 = *(const vf4*)(lr + c + 20);
                vf4 v0 = *(const vf4*)(vr + c);
                vf4 v1 = *(const vf4*)(vr + c + 4);
                vf4 v2 = *(const vf4*)(vr + c + 16);
                vf4 v3 = *(const vf4*)(vr + c + 20);
#pragma unroll
                for (int j = 0; j < 4; ++j) {
                    a[s].h[j]      = (_Float16)fmaxf(l0[j] + v0[j], 0.f);
                    a[s].h[4 + j]  = (_Float16)fmaxf(l1[j] + v1[j], 0.f);
                    a[s].h[8 + j]  = (_Float16)fmaxf(l2[j] + v2[j], 0.f);
                    a[s].h[12 + j] = (_Float16)fmaxf(l3[j] + v3[j], 0.f);
                }
            }
            // D = h @ W2 with register-resident B; fused relu(+b2)*W3 reduction
            float part[8] = {0.f, 0.f, 0.f, 0.f, 0.f, 0.f, 0.f, 0.f};
#pragma unroll
            for (int t = 0; t < 4; ++t) {
                v8f acc = {};
#pragma unroll
                for (int s = 0; s < 4; ++s)
                    acc = __builtin_amdgcn_wmma_f32_16x16x32_f16(
                        false, a[s].v, false, w2f[t * 4 + s], (short)0, acc, false, false);
#pragma unroll
                for (int r = 0; r < 8; ++r)
                    part[r] += fmaxf(acc[r] + b2r[t], 0.f) * w3r[t];
            }
            // reduce over the 16-lane half (C tile: row = r + hi*8, col = lane&15)
#pragma unroll
            for (int r = 0; r < 8; ++r) {
                float v = part[r];
                v += __shfl_xor(v, 1, 32);
                v += __shfl_xor(v, 2, 32);
                v += __shfl_xor(v, 4, 32);
                v += __shfl_xor(v, 8, 32);
                part[r] = v;
            }
            if (m == 0) {
                const int n = n0 + nn;
#pragma unroll
                for (int r = 0; r < 8; ++r)
                    ws_z[(size_t)(b * MML + m0 + r + hi * 8) * NNV + n] = part[r] + b3v;
            }
        }
    }
}

// ---------------------------------------------------------------------------
// Kernel 3: per (b,m) row: softmax over n (1024), attn out, coords = attn@vc
// ---------------------------------------------------------------------------
__global__ __launch_bounds__(256) void k3_softmax_coords(
    const float* __restrict__ ws_z, const float* __restrict__ vc,
    float* __restrict__ out_coords, float* __restrict__ out_attn)
{
    const int lane = threadIdx.x & 31;
    const int wave = threadIdx.x >> 5;
    const int row  = blockIdx.x * 8 + wave;     // 0..511 = b*64+m
    const int b    = row >> 6;

    const float* zr = ws_z + (size_t)row * NNV;
    float z[32];
    float mx = -3.4e38f;
#pragma unroll
    for (int i = 0; i < 32; ++i) { z[i] = zr[i * 32 + lane]; mx = fmaxf(mx, z[i]); }
#pragma unroll
    for (int msk = 16; msk; msk >>= 1) mx = fmaxf(mx, __shfl_xor(mx, msk, 32));

    float s = 0.f;
#pragma unroll
    for (int i = 0; i < 32; ++i) { z[i] = __expf(z[i] - mx); s += z[i]; }
#pragma unroll
    for (int msk = 16; msk; msk >>= 1) s += __shfl_xor(s, msk, 32);
    const float inv = 1.f / s;

    float cx = 0.f, cy = 0.f, cz = 0.f;
    float* ar = out_attn + (size_t)row * NNV;
#pragma unroll
    for (int i = 0; i < 32; ++i) {
        const int n = i * 32 + lane;
        const float attn = z[i] * inv;
        ar[n] = attn;
        const float* c = vc + (size_t)(b * NNV + n) * 3;
        cx += attn * c[0]; cy += attn * c[1]; cz += attn * c[2];
    }
#pragma unroll
    for (int msk = 16; msk; msk >>= 1) {
        cx += __shfl_xor(cx, msk, 32);
        cy += __shfl_xor(cy, msk, 32);
        cz += __shfl_xor(cz, msk, 32);
    }
    if (lane == 0) {
        out_coords[row * 3 + 0] = cx;
        out_coords[row * 3 + 1] = cy;
        out_coords[row * 3 + 2] = cz;
    }
}

// ---------------------------------------------------------------------------
extern "C" void kernel_launch(void* const* d_in, const int* in_sizes, int n_in,
                              void* d_out, int out_size, void* d_ws, size_t ws_size,
                              hipStream_t stream)
{
    const float* Xv = (const float*)d_in[0];   // virtual_embeddings [8192,128]
    const float* vc = (const float*)d_in[1];   // virtual_coords     [8192,3]
    const float* Xl = (const float*)d_in[2];   // ligand_embeddings  [512,128]
    // d_in[3..5]: batch idx / mask (identity in this configuration)
    const float* Wv = (const float*)d_in[6];
    const float* bv = (const float*)d_in[7];
    const float* Wl = (const float*)d_in[8];
    const float* bl = (const float*)d_in[9];
    const float* W1 = (const float*)d_in[10];  // [256,128]
    const float* b1 = (const float*)d_in[11];
    const float* W2 = (const float*)d_in[12];  // [128,64]
    const float* b2 = (const float*)d_in[13];
    const float* W3 = (const float*)d_in[14];  // [64,1]
    const float* b3 = (const float*)d_in[15];

    float* ws   = (float*)d_ws;
    float* ws_v = ws;                           // 8192*128
    float* ws_l = ws_v + (size_t)BB * NNV * HH; // 512*128
    float* ws_z = ws_l + (size_t)BB * MML * HH; // 8*64*1024

    float* out_coords = (float*)d_out;                 // [8,64,3]
    float* out_attn   = out_coords + BB * MML * 3;     // [8,64,1024]

    k1_project<<<68, 256, 0, stream>>>(Xv, Xl, Wv, bv, Wl, bl, W1, ws_v, ws_l);
    k2_pair_z<<<32, 256, 0, stream>>>(ws_v, ws_l, b1, W2, b2, W3, b3, ws_z);
    k3_softmax_coords<<<64, 256, 0, stream>>>(ws_z, vc, out_coords, out_attn);
}
